// StochasticConceptBottleneck_46170898432201
// MI455X (gfx1250) — compile-verified, matrix operand order checked by reading, not run on hardware
//
#include <hip/hip_runtime.h>
#include <math.h>

typedef __attribute__((ext_vector_type(16))) __bf16 v16bf;
typedef __attribute__((ext_vector_type(8)))  __bf16 v8bf;
typedef __attribute__((ext_vector_type(8)))  float  v8f;

static constexpr int CD   = 64;    // concepts
static constexpr int MC   = 100;   // monte carlo draws
static constexpr int MCP  = 112;   // MC padded to 7*16
static constexpr int TRI  = 2080;  // C*(C+1)/2
static constexpr int FD   = 512;   // features
static constexpr int BD   = 4096;  // batch
static constexpr int NCAT = CD + TRI;   // 2144
static constexpr int NPAD = 2176;       // 17*128

union V16 { v16bf v; v8bf h[2]; };

// ---------------- converts ----------------
__global__ void cvt_x_kernel(const float* __restrict__ x, __bf16* __restrict__ xb) {
  int i = blockIdx.x * blockDim.x + threadIdx.x;
  if (i < BD * FD) xb[i] = (__bf16)x[i];
}

__global__ void cvt_w_kernel(const float* __restrict__ wmu,
                             const float* __restrict__ wsig,
                             __bf16* __restrict__ wb) {
  int i = blockIdx.x * blockDim.x + threadIdx.x;
  if (i >= NPAD * FD) return;
  int row = i / FD, col = i - row * FD;
  float v = 0.0f;
  if (row < CD)        v = wmu[row * FD + col];
  else if (row < NCAT) v = wsig[(row - CD) * FD + col];
  wb[i] = (__bf16)v;
}

// ---------------- phase 1: Y = x @ [w_mu; w_sigma]^T + bias ----------------
__global__ __launch_bounds__(256)
void gemm_xw_kernel(const __bf16* __restrict__ A,   // [BD][FD] bf16
                    const __bf16* __restrict__ Bw,  // [NPAD][FD] bf16 (row n = weight row n)
                    const float* __restrict__ bmu,
                    const float* __restrict__ bsig,
                    float* __restrict__ Y)          // [BD][NPAD] f32
{
  const int lane = threadIdx.x & 31;
  const int wave = threadIdx.x >> 5;
  const int Mw = blockIdx.x * 64  + (wave & 1) * 32;
  const int Nw = blockIdx.y * 128 + (wave >> 1) * 32;

  const int lr = lane & 15;
  const int kh = (lane >> 4) * 8;        // K sub-offset per ISA A/B layout

  v8f acc00 = {}, acc01 = {}, acc10 = {}, acc11 = {};

  const __bf16* a0p = A  + (size_t)(Mw + lr)      * FD + kh;
  const __bf16* a1p = A  + (size_t)(Mw + 16 + lr) * FD + kh;
  const __bf16* b0p = Bw + (size_t)(Nw + lr)      * FD + kh;
  const __bf16* b1p = Bw + (size_t)(Nw + 16 + lr) * FD + kh;

  for (int kb = 0; kb < FD; kb += 32) {
    V16 a0, a1, b0, b1;
    a0.h[0] = *(const v8bf*)(a0p + kb);       a0.h[1] = *(const v8bf*)(a0p + kb + 16);
    a1.h[0] = *(const v8bf*)(a1p + kb);       a1.h[1] = *(const v8bf*)(a1p + kb + 16);
    b0.h[0] = *(const v8bf*)(b0p + kb);       b0.h[1] = *(const v8bf*)(b0p + kb + 16);
    b1.h[0] = *(const v8bf*)(b1p + kb);       b1.h[1] = *(const v8bf*)(b1p + kb + 16);
    acc00 = __builtin_amdgcn_wmma_f32_16x16x32_bf16(false, a0.v, false, b0.v, (short)0, acc00, false, false);
    acc01 = __builtin_amdgcn_wmma_f32_16x16x32_bf16(false, a0.v, false, b1.v, (short)0, acc01, false, false);
    acc10 = __builtin_amdgcn_wmma_f32_16x16x32_bf16(false, a1.v, false, b0.v, (short)0, acc10, false, false);
    acc11 = __builtin_amdgcn_wmma_f32_16x16x32_bf16(false, a1.v, false, b1.v, (short)0, acc11, false, false);
  }

  const int n0 = Nw + lr;
  const int n1 = n0 + 16;
  float bias0 = (n0 < CD) ? bmu[n0] : (n0 < NCAT ? bsig[n0 - CD] : 0.0f);
  float bias1 = (n1 < CD) ? bmu[n1] : (n1 < NCAT ? bsig[n1 - CD] : 0.0f);
  const int mbase = (lane >> 4) << 3;          // D layout: lanes>=16 hold M=r+8
  for (int r = 0; r < 8; ++r) {
    int m0 = Mw + mbase + r;
    int m1 = m0 + 16;
    Y[(size_t)m0 * NPAD + n0] = acc00[r] + bias0;
    Y[(size_t)m0 * NPAD + n1] = acc01[r] + bias1;
    Y[(size_t)m1 * NPAD + n0] = acc10[r] + bias0;
    Y[(size_t)m1 * NPAD + n1] = acc11[r] + bias1;
  }
}

// ---------------- phase 2: L build + mu + L@eps + sigmoid ----------------
__global__ __launch_bounds__(256)
void sample_kernel(const float* __restrict__ Y,    // [BD][NPAD]: col 0..63 = c_mu, 64.. = c_sigma
                   const float* __restrict__ eps,  // [BD][CD][MC]
                   float* __restrict__ out)        // [BD][CD][MC]
{
  __shared__ __align__(16) __bf16 Lsh[CD][72];     // 64x64 L, row-padded (stride 144B)
  __shared__ __align__(16) __bf16 Esh[MCP][72];    // eps^T: row = mc, col = concept k
  __shared__ float mu[CD];

  const int b    = blockIdx.x;
  const int tid  = threadIdx.x;
  const int lane = tid & 31;
  const int wave = tid >> 5;
  const float* yrow = Y + (size_t)b * NPAD;

  // zero L and Esh (upper triangle of L + MC padding rows must be 0)
  {
    unsigned int* p = (unsigned int*)&Lsh[0][0];
    for (int i = tid; i < (CD * 72) / 2; i += 256) p[i] = 0u;
    unsigned int* q = (unsigned int*)&Esh[0][0];
    for (int i = tid; i < (MCP * 72) / 2; i += 256) q[i] = 0u;
  }
  if (tid < CD) mu[tid] = yrow[tid];
  __syncthreads();

  // scatter c_sigma -> lower triangle; softplus(+1e-6) on the diagonal
  for (int t = tid; t < TRI; t += 256) {
    float s = yrow[CD + t];
    int i = (int)((sqrtf(8.0f * (float)t + 1.0f) - 1.0f) * 0.5f);
    while ((i + 1) * (i + 2) / 2 <= t) ++i;
    while (i * (i + 1) / 2 > t)       --i;
    int j = t - i * (i + 1) / 2;
    if (i == j) {
      s = (s > 20.0f) ? s : log1pf(__expf(s));
      s += 1e-6f;
    }
    Lsh[i][j] = (__bf16)s;
  }

  // eps[b] (row-major [CD][MC]) -> Esh transposed bf16
  const float* erow = eps + (size_t)b * (CD * MC);
  for (int e = tid; e < CD * MC; e += 256) {
    int c = e / MC;
    int m = e - c * MC;
    Esh[m][c] = (__bf16)erow[e];
  }
  __syncthreads();

  const int lr    = lane & 15;
  const int kh    = (lane >> 4) * 8;
  const int mbase = (lane >> 4) << 3;

  // 4 row-tiles x 7 col-tiles = 28 output tiles over 8 waves
  for (int tt = wave; tt < 28; tt += 8) {
    int rt = tt / 7;
    int ct = tt - rt * 7;

    v8f acc;
    for (int r = 0; r < 8; ++r) acc[r] = mu[rt * 16 + mbase + r];

    V16 a, bm;
    // K block 0..31
    a.h[0]  = *(const v8bf*)&Lsh[rt * 16 + lr][kh];
    a.h[1]  = *(const v8bf*)&Lsh[rt * 16 + lr][kh + 16];
    bm.h[0] = *(const v8bf*)&Esh[ct * 16 + lr][kh];
    bm.h[1] = *(const v8bf*)&Esh[ct * 16 + lr][kh + 16];
    acc = __builtin_amdgcn_wmma_f32_16x16x32_bf16(false, a.v, false, bm.v, (short)0, acc, false, false);
    // K block 32..63
    a.h[0]  = *(const v8bf*)&Lsh[rt * 16 + lr][32 + kh];
    a.h[1]  = *(const v8bf*)&Lsh[rt * 16 + lr][32 + kh + 16];
    bm.h[0] = *(const v8bf*)&Esh[ct * 16 + lr][32 + kh];
    bm.h[1] = *(const v8bf*)&Esh[ct * 16 + lr][32 + kh + 16];
    acc = __builtin_amdgcn_wmma_f32_16x16x32_bf16(false, a.v, false, bm.v, (short)0, acc, false, false);

    int mc = ct * 16 + lr;
    if (mc < MC) {
      float* orow = out + (size_t)b * (CD * MC) + mc;
      for (int r = 0; r < 8; ++r) {
        int cm = rt * 16 + mbase + r;
        float v = acc[r];
        orow[(size_t)cm * MC] = 1.0f / (1.0f + __expf(-v));
      }
    }
  }
}

// ---------------- launch ----------------
extern "C" void kernel_launch(void* const* d_in, const int* in_sizes, int n_in,
                              void* d_out, int out_size, void* d_ws, size_t ws_size,
                              hipStream_t stream) {
  const float* x     = (const float*)d_in[0];
  const float* eps   = (const float*)d_in[1];
  const float* w_mu  = (const float*)d_in[2];
  const float* b_mu  = (const float*)d_in[3];
  const float* w_sig = (const float*)d_in[4];
  const float* b_sig = (const float*)d_in[5];
  float* out = (float*)d_out;

  char* ws = (char*)d_ws;
  // ws layout (all 256B aligned):
  //   xb: BD*FD bf16      = 4,194,304 B
  //   wb: NPAD*FD bf16    = 2,228,224 B
  //   Y : BD*NPAD f32     = 35,651,584 B
  __bf16* xb = (__bf16*)ws;
  __bf16* wb = (__bf16*)(ws + 4194304);
  float*  Y  = (float*)(ws + 4194304 + 2228224);
  (void)ws_size; (void)in_sizes; (void)n_in; (void)out_size;

  {
    int n = BD * FD;
    cvt_x_kernel<<<(n + 255) / 256, 256, 0, stream>>>(x, xb);
  }
  {
    int n = NPAD * FD;
    cvt_w_kernel<<<(n + 255) / 256, 256, 0, stream>>>(w_mu, w_sig, wb);
  }
  gemm_xw_kernel<<<dim3(BD / 64, NPAD / 128), 256, 0, stream>>>(xb, wb, b_mu, b_sig, Y);
  sample_kernel<<<BD, 256, 0, stream>>>(Y, eps, out);
}
